// MultiHeadedAttention_8323646620320
// MI455X (gfx1250) — matrix operand order repaired and checked by reading, not compiled, not run on hardware
//
#include <hip/hip_runtime.h>

// ---------------------------------------------------------------------------
// MHA forward for MI455X (gfx1250, wave32, WMMA + Tensor Data Mover).
// Pipeline: cvt(f32->f16, weights transposed) -> GEMM(QKV, f16 out)
//           -> flash-attn (f16 in/out) -> GEMM(dense, f32 out).
// Matmul via v_wmma_f32_16x16x32_f16; LDS staging via tensor_load_to_lds.
// ---------------------------------------------------------------------------

typedef __attribute__((ext_vector_type(16))) _Float16     v16h;
typedef __attribute__((ext_vector_type(8)))  float        v8f;
typedef __attribute__((ext_vector_type(4)))  unsigned int u32x4;
typedef __attribute__((ext_vector_type(4)))  int          i32x4;
typedef __attribute__((ext_vector_type(8)))  int          i32x8;

#define HIDDEN 1024
#define HEADS  16
#define HEADD  64
#define SEQ    2048
#define BATCH  2

#if __has_builtin(__builtin_amdgcn_tensor_load_to_lds)
  #define HAVE_TDM 1
#else
  #define HAVE_TDM 0
#endif
#if defined(__has_include)
  #if __has_include(<hip/amd_detail/amd_gfx1250_TDM.h>)
    #define TDM_6ARG 1
  #else
    #define TDM_6ARG 0
  #endif
#else
  #define TDM_6ARG 0
#endif

union Frag { v16h v; uint4 q[2]; unsigned u[8]; };
union H8   { uint4 q; unsigned short h[8]; };

static __device__ inline unsigned pack2(float a, float b) {
    union { _Float16 h[2]; unsigned u; } p;
    p.h[0] = (_Float16)a; p.h[1] = (_Float16)b;
    return p.u;
}
static __device__ inline unsigned short f2h(float a) {
    union { _Float16 h; unsigned short u; } p;
    p.h = (_Float16)a; return p.u;
}
static __device__ inline v8f wmma_f16(const Frag& a, const Frag& b, v8f c) {
    return __builtin_amdgcn_wmma_f32_16x16x32_f16(false, a.v, false, b.v,
                                                  (short)0, c, false, false);
}

// ---------------------------------------------------------------------------
// TDM: DMA a 2D f16 tile [tile1 rows x tile0 halves], row stride `strideElems`
// halves, into LDS at byte offset ldsOff with per-row padding so the LDS pitch
// is tile0 + (padAmount+1)*2 halves.  padInterval: DWORDs/row = 2<<padInterval.
// Issued by one wave; completion via s_wait_tensorcnt.
// ---------------------------------------------------------------------------
#if HAVE_TDM
static __device__ inline void tdm_load_2d(unsigned ldsOff,
                                          const unsigned short* g,
                                          int tile0, int tile1,
                                          int strideElems,
                                          int padInterval, int padAmount) {
    unsigned long long ga = (unsigned long long)(size_t)g;
    u32x4 g0;
    g0[0] = 1u;                                        // count=1, gather off
    g0[1] = ldsOff;                                    // lds_addr (bytes)
    g0[2] = (unsigned)ga;                              // global_addr[31:0]
    g0[3] = (unsigned)((ga >> 32) & 0x01FFFFFFu)       // global_addr[56:32]
          | 0x80000000u;                               // type=2 ("image")
    i32x8 g1;
    g1[0] = (int)((1u << 16)                           // data_size = 2 bytes
          | (1u << 20)                                 // pad_enable
          | ((unsigned)padInterval << 22)
          | ((unsigned)padAmount   << 25));
    g1[1] = (int)(((unsigned)tile0 & 0xFFFFu) << 16);          // tensor_dim0 lo
    g1[2] = (int)((((unsigned)tile0 >> 16) & 0xFFFFu)          // tensor_dim0 hi
          | (((unsigned)tile1 & 0xFFFFu) << 16));              // tensor_dim1 lo
    g1[3] = (int)((((unsigned)tile1 >> 16) & 0xFFFFu)          // tensor_dim1 hi
          | (((unsigned)tile0 & 0xFFFFu) << 16));              // tile_dim0
    g1[4] = (int)((unsigned)tile1 & 0xFFFFu);                  // tile_dim1
    g1[5] = strideElems;                                       // dim0_stride lo
    g1[6] = 0;
    g1[7] = 0;
    i32x4 z4 = {0, 0, 0, 0};
#if TDM_6ARG
    i32x8 z8 = {0, 0, 0, 0, 0, 0, 0, 0};
    __builtin_amdgcn_tensor_load_to_lds(g0, g1, z4, z4, z8, 0);
#else
    __builtin_amdgcn_tensor_load_to_lds(g0, g1, z4, z4, 0);
#endif
}
static __device__ inline void tdm_wait() {
    __builtin_amdgcn_s_wait_tensorcnt(0);
}
#endif

// ---------------------------------------------------------------------------
// One-shot f32 -> f16 conversion (element pairs)
// ---------------------------------------------------------------------------
__global__ __launch_bounds__(256)
void cvt_f16(const float* __restrict__ src, unsigned short* __restrict__ dst,
             int npairs) {
    int i = blockIdx.x * 256 + threadIdx.x;
    if (i < npairs) {
        float2 v = ((const float2*)src)[i];
        ((unsigned*)dst)[i] = pack2(v.x, v.y);
    }
}

// One-shot f32 [K,N] -> f16 [N,K] transpose (32x32 LDS tiles, 256 threads)
__global__ __launch_bounds__(256)
void cvt_f16_t(const float* __restrict__ src, unsigned short* __restrict__ dst,
               int K, int N) {
    __shared__ float tile[32][33];
    const int k0 = blockIdx.y * 32;
    const int n0 = blockIdx.x * 32;
    const int tx = threadIdx.x & 31;
    const int ty = threadIdx.x >> 5;           // 32x8
    #pragma unroll
    for (int r = 0; r < 4; ++r)
        tile[ty + r * 8][tx] = src[(size_t)(k0 + ty + r * 8) * N + n0 + tx];
    __syncthreads();
    #pragma unroll
    for (int r = 0; r < 4; ++r)
        dst[(size_t)(n0 + ty + r * 8) * K + k0 + tx] = f2h(tile[tx][ty + r * 8]);
}

// ---------------------------------------------------------------------------
// GEMM: C[M,N] = A[M,K]*Bt[N,K]^T + bias[N].  A,Bt f16 row-major; C f16/f32.
// 256 threads = 8 waves; 128x128 tile, K-step 32, double-buffered LDS,
// staged by the Tensor Data Mover (wave 0), WMMA overlapped with DMA.
// ---------------------------------------------------------------------------
#define LP 40  // LDS pitch in halves = 32 + 8 pad (TDM: interval 16 DW, pad 4 DW)

template <bool OUT_F16>
__global__ __launch_bounds__(256)
void gemm_f16wmma(const unsigned short* __restrict__ A,
                  const unsigned short* __restrict__ Bt,
                  const float* __restrict__ bias, void* __restrict__ Cv,
                  int M, int N, int K) {
    __shared__ __align__(16) unsigned short lA[2][128 * LP];   // [m][k]
    __shared__ __align__(16) unsigned short lB[2][128 * LP];   // [n][k]

    const int tid  = threadIdx.x;
    const int lane = tid & 31;
    const int wave = tid >> 5;
    const int wr   = wave >> 1;
    const int wc   = wave & 1;
    const int hi   = lane >> 4;
    const int l15  = lane & 15;
    const int mBase = blockIdx.y * 128;
    const int nBase = blockIdx.x * 128;
    const int ntiles = K / 32;

    // Per-thread coords for fallback copy / prefetch (2 uint4 each tile)
    int sRow[2], sCol[2];
    #pragma unroll
    for (int r = 0; r < 2; ++r) {
        int flat = tid + r * 256;          // 512 uint4 per 128x32-half tile
        sRow[r] = flat >> 2;
        sCol[r] = (flat & 3) * 8;
    }

    v8f acc[2][4];
    #pragma unroll
    for (int i = 0; i < 2; ++i)
        #pragma unroll
        for (int j = 0; j < 4; ++j) acc[i][j] = {};

    const unsigned ldsA[2] = { (unsigned)(size_t)&lA[0][0], (unsigned)(size_t)&lA[1][0] };
    const unsigned ldsB[2] = { (unsigned)(size_t)&lB[0][0], (unsigned)(size_t)&lB[1][0] };
    const unsigned short* Ab = A  + (size_t)mBase * K;
    const unsigned short* Bb = Bt + (size_t)nBase * K;

    // Stage tile 0 into buffer 0
#if HAVE_TDM
    if (wave == 0) {
        tdm_load_2d(ldsA[0], Ab, 32, 128, K, 3, 3);
        tdm_load_2d(ldsB[0], Bb, 32, 128, K, 3, 3);
    }
#else
    #pragma unroll
    for (int r = 0; r < 2; ++r) {
        *(uint4*)&lA[0][sRow[r] * LP + sCol[r]] = *(const uint4*)(Ab + (size_t)sRow[r] * K + sCol[r]);
        *(uint4*)&lB[0][sRow[r] * LP + sCol[r]] = *(const uint4*)(Bb + (size_t)sRow[r] * K + sCol[r]);
    }
#endif

    for (int it = 0; it < ntiles; ++it) {
        const int cur = it & 1;
#if HAVE_TDM
        if (wave == 0) tdm_wait();         // DMA for buf[cur] complete
#endif
        __syncthreads();                   // buf[cur] visible; buf[cur^1] free

        if (it + 1 < ntiles) {             // stage next tile into buf[cur^1]
            const int k0n = (it + 1) * 32;
#if HAVE_TDM
            if (wave == 0) {
                tdm_load_2d(ldsA[cur ^ 1], Ab + k0n, 32, 128, K, 3, 3);
                tdm_load_2d(ldsB[cur ^ 1], Bb + k0n, 32, 128, K, 3, 3);
            }
#else
            #pragma unroll
            for (int r = 0; r < 2; ++r) {
                *(uint4*)&lA[cur ^ 1][sRow[r] * LP + sCol[r]] =
                    *(const uint4*)(Ab + (size_t)sRow[r] * K + k0n + sCol[r]);
                *(uint4*)&lB[cur ^ 1][sRow[r] * LP + sCol[r]] =
                    *(const uint4*)(Bb + (size_t)sRow[r] * K + k0n + sCol[r]);
            }
#endif
            if (it + 2 < ntiles) {         // warm L2/WGP$ for tile after next
                const int kp = (it + 2) * 32;
                __builtin_prefetch(Ab + (size_t)sRow[0] * K + kp + sCol[0], 0, 1);
                __builtin_prefetch(Bb + (size_t)sRow[0] * K + kp + sCol[0], 0, 1);
            }
        }

        // Fragments from LDS buffer `cur` and 8 WMMAs
        Frag af[2], bf[4];
        #pragma unroll
        for (int mt = 0; mt < 2; ++mt) {
            int row = wr * 32 + mt * 16 + l15;
            int kb  = hi * 8;   // A layout: halves 0-7 -> k=kb.., 8-15 -> kb+16..
            af[mt].q[0] = *(const uint4*)&lA[cur][row * LP + kb];
            af[mt].q[1] = *(const uint4*)&lA[cur][row * LP + kb + 16];
        }
        #pragma unroll
        for (int nt = 0; nt < 4; ++nt) {
            int n  = wc * 64 + nt * 16 + l15;
            int kb = hi * 8;    // B layout: halves 0-7 -> k=kb.., 8-15 -> kb+16..
            bf[nt].q[0] = *(const uint4*)&lB[cur][n * LP + kb];
            bf[nt].q[1] = *(const uint4*)&lB[cur][n * LP + kb + 16];
        }
        #pragma unroll
        for (int mt = 0; mt < 2; ++mt)
            #pragma unroll
            for (int nt = 0; nt < 4; ++nt)
                acc[mt][nt] = wmma_f16(af[mt], bf[nt], acc[mt][nt]);
    }

    // Epilogue: C layout VGPR r -> row r + hi*8, lane -> n
    #pragma unroll
    for (int mt = 0; mt < 2; ++mt)
        #pragma unroll
        for (int nt = 0; nt < 4; ++nt) {
            int   n  = nBase + wc * 64 + nt * 16 + l15;
            float bs = bias[n];
            int   m0 = mBase + wr * 32 + mt * 16 + hi * 8;
            if constexpr (OUT_F16) {
                unsigned short* cp = (unsigned short*)Cv + (size_t)m0 * N + n;
                #pragma unroll
                for (int r = 0; r < 8; ++r)
                    cp[(size_t)r * N] = f2h(acc[mt][nt][r] + bs);
            } else {
                float* cp = (float*)Cv + (size_t)m0 * N + n;
                #pragma unroll
                for (int r = 0; r < 8; ++r)
                    cp[(size_t)r * N] = acc[mt][nt][r] + bs;
            }
        }
}

// ---------------------------------------------------------------------------
// Flash attention on f16 qkv: block = (batch, head, 64-query block),
// 128 threads = 4 waves, each wave owns 16 query rows; 32-key chunks.
// K tile DMA'd by TDM; V^T transposed through registers.
// ---------------------------------------------------------------------------
__global__ __launch_bounds__(128)
void attn_f16wmma(const unsigned short* __restrict__ qkv,
                  unsigned short* __restrict__ ctxOut) {
    __shared__ __align__(16) unsigned short lK[32 * 72];      // [key][d], pitch 72
    __shared__ __align__(16) unsigned short lVT[64 * 40];     // [d][key], pitch 40
    __shared__ __align__(16) unsigned short lP[4 * 16 * 40];  // per-wave P scratch

    const int tid  = threadIdx.x;
    const int lane = tid & 31;
    const int wave = tid >> 5;
    const int hi   = lane >> 4;
    const int l15  = lane & 15;

    const int bid   = blockIdx.x;
    const int qb    = bid & 31;
    const int head  = (bid >> 5) & 15;
    const int batch = bid >> 9;

    const int rowStride = 3 * HIDDEN;
    const int qcol = head * 3 * HEADD;
    const int kcol = qcol + HEADD;
    const int vcol = qcol + 2 * HEADD;

    // Q as two A-frags (d chunks 0-31 / 32-63), straight from global f16.
    Frag qf[2];
    {
        int srow = batch * SEQ + qb * 64 + wave * 16 + l15;
        const unsigned short* qp = qkv + (size_t)srow * rowStride + qcol;
        int kb = hi * 8;
        #pragma unroll
        for (int f = 0; f < 2; ++f) {
            qf[f].q[0] = *(const uint4*)(qp + f * 32 + kb);
            qf[f].q[1] = *(const uint4*)(qp + f * 32 + kb + 16);
        }
    }

    v8f cacc[4];
    #pragma unroll
    for (int nt = 0; nt < 4; ++nt) cacc[nt] = {};
    float mrun[8], lrun[8];
    #pragma unroll
    for (int r = 0; r < 8; ++r) { mrun[r] = -1e30f; lrun[r] = 0.0f; }

    const unsigned ldsK = (unsigned)(size_t)&lK[0];

    for (int kb0 = 0; kb0 < SEQ; kb0 += 32) {
        __syncthreads();                   // previous chunk done reading LDS
        const unsigned short* kvBase =
            qkv + (size_t)(batch * SEQ + kb0) * rowStride;
#if HAVE_TDM
        if (wave == 0)                     // K tile: 32 keys x 64 d, pitch 72
            tdm_load_2d(ldsK, kvBase + kcol, 64, 32, rowStride, 4, 3);
#endif
        // V^T [d][key] via register transpose (and K copy on fallback)
        #pragma unroll
        for (int r = 0; r < 2; ++r) {
            int flat = tid + r * 128;      // 256 uint4 per 32x64-half tile
            int key  = flat >> 3;
            int q4   = (flat & 7) * 8;
            const unsigned short* kvRow = kvBase + (size_t)key * rowStride;
#if !HAVE_TDM
            *(uint4*)&lK[key * 72 + q4] = *(const uint4*)(kvRow + kcol + q4);
#endif
            H8 vv; vv.q = *(const uint4*)(kvRow + vcol + q4);
            #pragma unroll
            for (int j = 0; j < 8; ++j)
                lVT[(q4 + j) * 40 + key] = vv.h[j];
        }
#if HAVE_TDM
        if (wave == 0) tdm_wait();
#endif
        __syncthreads();

        // Scores: two 16-key tiles, contraction over d=64 (two k32 steps)
        v8f s[2];
        #pragma unroll
        for (int t = 0; t < 2; ++t) {
            s[t] = {};
            #pragma unroll
            for (int dc = 0; dc < 2; ++dc) {
                Frag kf;
                int off = (t * 16 + l15) * 72 + dc * 32 + hi * 8;
                kf.q[0] = *(const uint4*)&lK[off];
                kf.q[1] = *(const uint4*)&lK[off + 16];
                s[t] = wmma_f16(qf[dc], kf, s[t]);
            }
        }

        // Online softmax (row m = r + hi*8 lives on a 16-lane half-wave)
        float p0[8], p1[8];
        #pragma unroll
        for (int r = 0; r < 8; ++r) {
            float a0 = s[0][r] * 0.125f;   // 1/sqrt(64)
            float a1 = s[1][r] * 0.125f;
            float mx = fmaxf(a0, a1);
            #pragma unroll
            for (int off = 8; off > 0; off >>= 1)
                mx = fmaxf(mx, __shfl_xor(mx, off, 16));
            float mnew = fmaxf(mrun[r], mx);
            float corr = __expf(mrun[r] - mnew);
            mrun[r] = mnew;
            float e0 = __expf(a0 - mnew);
            float e1 = __expf(a1 - mnew);
            float sum = e0 + e1;
            #pragma unroll
            for (int off = 8; off > 0; off >>= 1)
                sum += __shfl_xor(sum, off, 16);
            lrun[r] = lrun[r] * corr + sum;
            #pragma unroll
            for (int nt = 0; nt < 4; ++nt) cacc[nt][r] *= corr;
            p0[r] = e0; p1[r] = e1;
        }

        // C-layout -> A-layout via per-wave LDS scratch (same-wave LDS in order)
        unsigned short* pw = &lP[wave * 16 * 40];
        #pragma unroll
        for (int r = 0; r < 8; ++r) {
            int mrow = r + hi * 8;
            pw[mrow * 40 + l15]      = f2h(p0[r]);
            pw[mrow * 40 + 16 + l15] = f2h(p1[r]);
        }
        Frag pf;
        {
            int kb = hi * 8;
            pf.q[0] = *(const uint4*)&pw[l15 * 40 + kb];
            pf.q[1] = *(const uint4*)&pw[l15 * 40 + kb + 16];
        }

        // ctx += P * V : B-frag from V^T (lane -> d column, halves -> keys)
        #pragma unroll
        for (int nt = 0; nt < 4; ++nt) {
            Frag vf;
            int off = (nt * 16 + l15) * 40 + hi * 8;
            vf.q[0] = *(const uint4*)&lVT[off];
            vf.q[1] = *(const uint4*)&lVT[off + 16];
            cacc[nt] = wmma_f16(pf, vf, cacc[nt]);
        }
    }

    // ctx / rowsum -> ctx16[B*S, HIDDEN] at columns head*64 + d
    #pragma unroll
    for (int nt = 0; nt < 4; ++nt) {
        int col = head * HEADD + nt * 16 + l15;
        #pragma unroll
        for (int r = 0; r < 8; ++r) {
            int mrow = batch * SEQ + qb * 64 + wave * 16 + r + hi * 8;
            ctxOut[(size_t)mrow * HIDDEN + col] = f2h(cacc[nt][r] / lrun[r]);
        }
    }
}

// ---------------------------------------------------------------------------
extern "C" void kernel_launch(void* const* d_in, const int* in_sizes, int n_in,
                              void* d_out, int out_size, void* d_ws, size_t ws_size,
                              hipStream_t stream) {
    const float* hs      = (const float*)d_in[0];   // [2,2048,1024]
    const float* w_qkv   = (const float*)d_in[1];   // [1024,3072]
    const float* b_qkv   = (const float*)d_in[2];   // [3072]
    const float* w_dense = (const float*)d_in[3];   // [1024,1024]
    const float* b_dense = (const float*)d_in[4];   // [1024]
    float* out = (float*)d_out;                     // [2,2048,1024]

    const int M = BATCH * SEQ;                      // 4096
    unsigned short* hs16   = (unsigned short*)d_ws;               // 4096*1024
    unsigned short* wq16t  = hs16   + (size_t)M * HIDDEN;         // [3072,1024]
    unsigned short* wd16t  = wq16t  + (size_t)3 * HIDDEN * HIDDEN;// [1024,1024]
    unsigned short* qkv16  = wd16t  + (size_t)HIDDEN * HIDDEN;    // 4096*3072
    unsigned short* ctx16  = qkv16  + (size_t)M * 3 * HIDDEN;     // 4096*1024

    // f32 -> f16 (activations) and f32 [K,N] -> f16 [N,K] (weights)
    {
        int np = M * HIDDEN / 2;
        cvt_f16<<<dim3(np / 256), dim3(256), 0, stream>>>(hs, hs16, np);
        cvt_f16_t<<<dim3(3 * HIDDEN / 32, HIDDEN / 32), dim3(256), 0, stream>>>(
            w_qkv, wq16t, HIDDEN, 3 * HIDDEN);
        cvt_f16_t<<<dim3(HIDDEN / 32, HIDDEN / 32), dim3(256), 0, stream>>>(
            w_dense, wd16t, HIDDEN, HIDDEN);
    }

    // QKV projection: [4096,1024] x [1024,3072] + bias -> f16
    gemm_f16wmma<true><<<dim3(3 * HIDDEN / 128, M / 128), dim3(256), 0, stream>>>(
        hs16, wq16t, b_qkv, qkv16, M, 3 * HIDDEN, HIDDEN);

    // Flash attention: 2 batches x 16 heads x 32 query-blocks -> f16 ctx
    attn_f16wmma<<<dim3(BATCH * HEADS * (SEQ / 64)), dim3(128), 0, stream>>>(
        qkv16, ctx16);

    // Output projection: [4096,1024] x [1024,1024] + bias -> f32
    gemm_f16wmma<false><<<dim3(HIDDEN / 128, M / 128), dim3(256), 0, stream>>>(
        ctx16, wd16t, b_dense, out, M, HIDDEN, HIDDEN);
}
